// Net_60026462929062
// MI455X (gfx1250) — matrix operand order repaired and checked by reading, not compiled, not run on hardware
//
#include <hip/hip_runtime.h>
#include <hip/hip_bf16.h>

typedef __bf16 bf16;
typedef __attribute__((ext_vector_type(16))) __bf16 v16bf;
typedef __attribute__((ext_vector_type(8)))  float  v8f;
typedef __attribute__((ext_vector_type(2)))  float  v2f;
typedef unsigned int u32x4 __attribute__((ext_vector_type(4)));
typedef unsigned int u32x8 __attribute__((ext_vector_type(8)));

#define NDIM 8192   // graph nodes (GEMM M and K)

// ---------------- A fp32 -> bf16 (128 MB, fits the 192 MB L2) ----------------
__global__ void k_cvt_a(const float* __restrict__ a, bf16* __restrict__ o) {
  size_t i = ((size_t)blockIdx.x * 256 + threadIdx.x) * 4;
  const float4 v = *(const float4*)(a + i);
  o[i + 0] = (bf16)v.x; o[i + 1] = (bf16)v.y;
  o[i + 2] = (bf16)v.z; o[i + 3] = (bf16)v.w;
}

// ---- x [B,N,2] -> T0 [n][b*2+f], fp32 + bf16 ----
__global__ void k_prep_x(const float* __restrict__ x, float* __restrict__ t32,
                         bf16* __restrict__ t16) {
  int i = blockIdx.x * 256 + threadIdx.x;   // 16*8192*2
  int b = i >> 14;
  int rem = i & 16383;
  int n = rem >> 1, f = rem & 1;
  float v = x[i];
  int o = n * 32 + b * 2 + f;
  t32[o] = v; t16[o] = (bf16)v;
}

// ---- OutAcc[n][b*32+c] (+)= sum_f T[n][b*FI+f] * w[f*32+c] ----
template <int FI>
__global__ void k_mix(const float* __restrict__ T, const float* __restrict__ w,
                      float* __restrict__ acc, int init) {
  int i = blockIdx.x * 256 + threadIdx.x;   // N*16*32
  int c = i & 31;
  int nb = i >> 5;
  int b = nb & 15;
  int n = nb >> 4;
  const float* t = T + (size_t)n * (16 * FI) + b * FI;
  float s = 0.f;
#pragma unroll
  for (int f = 0; f < FI; ++f) s += t[f] * w[f * 32 + c];
  size_t o = (size_t)n * 512 + b * 32 + c;
  if (init) acc[o] = s; else acc[o] += s;
}

// ---------------- core: O = alpha*(A @ X) + beta*Tprev (Chebyshev fused) ----------------
// A tile (BM x 64 bf16, row stride NDIM) is DMAed into LDS by the Tensor Data
// Mover with hardware row padding; X tile is manually transposed into LDS.
// Double-buffered, TENSORcnt-pipelined.
template <int BM, int BN, int WAVES_M, int WAVES_N, int C>
__global__ __launch_bounds__(256) void k_gemm_cheb(
    const bf16* __restrict__ A, const bf16* __restrict__ X,
    const float* __restrict__ Tprev, float alpha, float beta,
    float* __restrict__ O32, bf16* __restrict__ O16) {
  constexpr int KT  = 64;
  constexpr int LDA = KT + 16;          // 80 halfwords = 160B padded rows
  constexpr int WM  = BM / WAVES_M;
  constexpr int WN  = BN / WAVES_N;
  constexpr int MT  = WM / 16;
  constexpr int NT  = WN / 16;
  constexpr int NIT = NDIM / KT;

  __shared__ alignas(16) bf16 As[2][BM][LDA];
  __shared__ alignas(16) bf16 Xs[2][BN][LDA];   // transposed: Xs[col][k]

  const int tid   = threadIdx.x;
  const int lane  = tid & 31;
  const int wid   = tid >> 5;
  const int waveM = wid / WAVES_N;
  const int waveN = wid % WAVES_N;
  const int rowBase = blockIdx.y * BM;
  const int colBase = blockIdx.x * BN;

  v8f acc[MT][NT] = {};

  const int hi    = lane >> 4;   // lane half
  const int l15   = lane & 15;
  const int koffA = hi * 8;      // A frag halfword offset (K 0..7 / 8..15)
  const int koffB = hi * 8;      // B frag halfword offset (K 0..7 / 8..15)

  // ---- TDM: DMA one BM x KT bf16 tile (row stride NDIM) into padded LDS ----
  auto tdm_issue_a = [&](int buf, int k0) {
    unsigned long long ga =
        (unsigned long long)(uintptr_t)(A + (size_t)rowBase * NDIM + k0);
    unsigned lds = (unsigned)(uintptr_t)(&As[buf][0][0]);
    u32x4 g0;
    g0[0] = 1u;                                     // count=1, user descriptor
    g0[1] = lds;                                    // lds_addr
    g0[2] = (unsigned)ga;                           // global_addr[31:0]
    g0[3] = (unsigned)((ga >> 32) & 0x01ffffffull)  // global_addr[56:32]
            | (2u << 30);                           // type=2 (image)
    u32x8 g1;
    g1[0] = (1u << 16)        // data_size = 1 (2 bytes)
          | (1u << 20)        // pad_enable
          | (4u << 22)        // pad_interval: 32 DWORDs (one 128B row)
          | (7u << 25);       // pad_amount: 8 DWORDs (32B) -> 160B LDS rows
    g1[1] = ((unsigned)KT) << 16;                   // tensor_dim0 = 64
    g1[2] = ((unsigned)BM) << 16;                   // tensor_dim1 = 128
    g1[3] = ((unsigned)KT) << 16;                   // tile_dim0 = 64
    g1[4] = (unsigned)BM;                           // tile_dim1 = 128
    g1[5] = (unsigned)NDIM;                         // tensor_dim0_stride[31:0]
    g1[6] = 0u;
    g1[7] = 0u;
    asm volatile("tensor_load_to_lds %0, %1" :: "s"(g0), "s"(g1) : "memory");
  };

  // ---- manual transposed fill of the X tile (TDM cannot transpose) ----
  auto fill_x = [&](int buf, int k0) {
    constexpr int DWR = BN / 2;          // dwords per tile row
    constexpr int NDW = KT * DWR;        // dwords per tile
#pragma unroll
    for (int e = 0; e < NDW / 256; ++e) {
      int id = tid + e * 256;
      int k = id / DWR, cp = id % DWR;
      int c = cp * 2;
      unsigned d = *(const unsigned*)(X + (size_t)(k0 + k) * C + colBase + c);
      Xs[buf][c][k]     = __builtin_bit_cast(bf16, (unsigned short)(d & 0xffffu));
      Xs[buf][c + 1][k] = __builtin_bit_cast(bf16, (unsigned short)(d >> 16));
    }
  };

  if (wid == 0) tdm_issue_a(0, 0);
  fill_x(0, 0);

  for (int kt = 0; kt < NIT; ++kt) {
    const int cur = kt & 1;
    if (kt + 1 < NIT) {
      if (wid == 0) tdm_issue_a(cur ^ 1, (kt + 1) * KT);   // prefetch next tile
      fill_x(cur ^ 1, (kt + 1) * KT);
      if (wid == 0) __builtin_amdgcn_s_wait_tensorcnt(1);  // current tile done
    } else {
      if (wid == 0) __builtin_amdgcn_s_wait_tensorcnt(0);
    }
    __syncthreads();   // publish TDM + ds writes of current buffer

#pragma unroll
    for (int ks = 0; ks < KT; ks += 32) {
      v16bf af[MT], bfm[NT];
#pragma unroll
      for (int mt = 0; mt < MT; ++mt) {
        const bf16* p = &As[cur][waveM * WM + mt * 16 + l15][ks];
        union { v16bf v; u32x4 q[2]; } u;
        u.q[0] = *(const u32x4*)(p + koffA);
        u.q[1] = *(const u32x4*)(p + 16 + koffA);
        af[mt] = u.v;
      }
#pragma unroll
      for (int nt = 0; nt < NT; ++nt) {
        const bf16* p = &Xs[cur][waveN * WN + nt * 16 + l15][ks];
        union { v16bf v; u32x4 q[2]; } u;
        u.q[0] = *(const u32x4*)(p + koffB);
        u.q[1] = *(const u32x4*)(p + 16 + koffB);
        bfm[nt] = u.v;
      }
#pragma unroll
      for (int mt = 0; mt < MT; ++mt)
#pragma unroll
        for (int nt = 0; nt < NT; ++nt)
          acc[mt][nt] = __builtin_amdgcn_wmma_f32_16x16x32_bf16(
              false, af[mt], false, bfm[nt], (short)0, acc[mt][nt], false, false);
    }
    __syncthreads();   // current buffer fully consumed before it is refilled
  }

  // fused Chebyshev epilogue: t = alpha*P + beta*Tprev ; write fp32 + bf16
#pragma unroll
  for (int mt = 0; mt < MT; ++mt) {
#pragma unroll
    for (int nt = 0; nt < NT; ++nt) {
      const int col = colBase + waveN * WN + nt * 16 + l15;
#pragma unroll
      for (int v = 0; v < 8; ++v) {
        const int row = rowBase + waveM * WM + mt * 16 + v + hi * 8;
        const size_t o = (size_t)row * C + col;
        float t = alpha * acc[mt][nt][v];
        if (beta != 0.f) t += beta * Tprev[o];
        O32[o] = t;
        O16[o] = (bf16)t;
      }
    }
  }
}

// ---- bias + relu (optionally also produce bf16 copy for next layer's GEMM input) ----
__global__ void k_bias_relu(const float* __restrict__ acc, const float* __restrict__ bias,
                            float* __restrict__ o32, bf16* __restrict__ o16) {
  int i = blockIdx.x * 256 + threadIdx.x;   // N*512
  float v = acc[i] + bias[i & 31];
  v = fmaxf(v, 0.f);
  o32[i] = v;
  if (o16) o16[i] = (bf16)v;
}

// ---- fc1: [16 x 262144] @ [262144 x 512] in fp32 via V_WMMA_F32_16X16X4_F32, split-K=64 ----
__global__ __launch_bounds__(32) void k_fc1_wmma(const float* __restrict__ H,
                                                 const float* __restrict__ fw1,
                                                 float* __restrict__ part) {
  const int lane = threadIdx.x;
  const int m15  = lane & 15;
  const int hi   = lane >> 4;
  const int koff = hi * 2;                  // A 16x4 f32: lanes 0-15 K{0,1}, 16-31 K{2,3}
  const int colBase = blockIdx.x * 16;
  const int kBase = blockIdx.y * 4096;
  v8f acc = {};
  for (int k0 = 0; k0 < 4096; k0 += 4) {
    const int k = kBase + k0 + koff;
    // h[b][k] with k = n*32+c lives at H[n*512 + b*32 + c]
    const float* ap = H + (size_t)(k >> 5) * 512 + m15 * 32 + (k & 31);
    v2f av; av.x = ap[0]; av.y = ap[1];
    const float* bp = fw1 + (size_t)k * 512 + colBase + m15;
    v2f bv; bv.x = bp[0]; bv.y = bp[512];
    acc = __builtin_amdgcn_wmma_f32_16x16x4_f32(
        false, av, false, bv, (short)0, acc, false, false);
  }
#pragma unroll
  for (int v = 0; v < 8; ++v) {
    const int row = v + hi * 8;             // batch index
    part[(size_t)blockIdx.y * 8192 + row * 512 + colBase + m15] = acc[v];
  }
}

__global__ void k_fc1_reduce(const float* __restrict__ part, const float* __restrict__ fb1,
                             float* __restrict__ o) {
  int i = blockIdx.x * 256 + threadIdx.x;   // 16*512
  float s = fb1[i & 511];
  for (int sp = 0; sp < 64; ++sp) s += part[(size_t)sp * 8192 + i];
  o[i] = fmaxf(s, 0.f);
}

__global__ void k_fc2(const float* __restrict__ h, const float* __restrict__ fw2,
                      const float* __restrict__ fb2, float* __restrict__ o) {
  int i = blockIdx.x * 256 + threadIdx.x;   // 16*128
  int b = i >> 7, m = i & 127;
  float s = fb2[m];
  for (int j = 0; j < 512; ++j) s += h[b * 512 + j] * fw2[j * 128 + m];
  o[i] = fmaxf(s, 0.f);
}

__global__ void k_fc3_softmax(const float* __restrict__ h, const float* __restrict__ fw3,
                              const float* __restrict__ fb3, float* __restrict__ out) {
  int b = threadIdx.x;
  if (b >= 16) return;
  float s0 = fb3[0], s1 = fb3[1];
  for (int j = 0; j < 128; ++j) {
    float v = h[b * 128 + j];
    s0 += v * fw3[j * 2 + 0];
    s1 += v * fw3[j * 2 + 1];
  }
  float mx = fmaxf(s0, s1);
  float e0 = expf(s0 - mx), e1 = expf(s1 - mx);
  float inv = 1.f / (e0 + e1);
  out[b * 2 + 0] = e0 * inv;
  out[b * 2 + 1] = e1 * inv;
}

extern "C" void kernel_launch(void* const* d_in, const int* in_sizes, int n_in,
                              void* d_out, int out_size, void* d_ws, size_t ws_size,
                              hipStream_t stream) {
  const float* x   = (const float*)d_in[0];
  const float* a   = (const float*)d_in[1];
  const float* w1  = (const float*)d_in[2];
  const float* b1  = (const float*)d_in[3];
  const float* w2  = (const float*)d_in[4];
  const float* b2  = (const float*)d_in[5];
  const float* fw1 = (const float*)d_in[6];
  const float* fb1 = (const float*)d_in[7];
  const float* fw2 = (const float*)d_in[8];
  const float* fb2 = (const float*)d_in[9];
  const float* fw3 = (const float*)d_in[10];
  const float* fb3 = (const float*)d_in[11];
  float* out = (float*)d_out;
  (void)in_sizes; (void)n_in; (void)out_size; (void)ws_size;

  char* ws = (char*)d_ws;
  size_t off = 0;
  auto alloc = [&](size_t bytes) -> void* {
    void* p = ws + off;
    off += (bytes + 255) & ~(size_t)255;
    return p;
  };
  const size_t NN = (size_t)NDIM * NDIM;
  bf16* A16 = (bf16*)alloc(NN * 2);                                  // 128 MB
  float* P[3];
  for (int i = 0; i < 3; ++i) P[i] = (float*)alloc((size_t)NDIM * 512 * 4);
  bf16* Q[2];
  for (int i = 0; i < 2; ++i) Q[i] = (bf16*)alloc((size_t)NDIM * 512 * 2);
  float* Acc  = (float*)alloc((size_t)NDIM * 512 * 4);
  float* part = (float*)alloc((size_t)64 * 16 * 512 * 4);
  float* f1o  = (float*)alloc((size_t)16 * 512 * 4);
  float* f2o  = (float*)alloc((size_t)16 * 128 * 4);

  k_cvt_a<<<NN / (256 * 4), 256, 0, stream>>>(a, A16);
  k_prep_x<<<(16 * NDIM * 2) / 256, 256, 0, stream>>>(x, P[0], Q[0]);

  const int mixBlocks = (NDIM * 16 * 32) / 256;

  // ---------------- layer 1 (F_in = 2, C = 32) ----------------
  k_mix<2><<<mixBlocks, 256, 0, stream>>>(P[0], w1, Acc, 1);
  k_gemm_cheb<128, 32, 8, 1, 32><<<dim3(1, 64), 256, 0, stream>>>(
      A16, Q[0], nullptr, 1.f, 0.f, P[1], Q[1]);
  k_mix<2><<<mixBlocks, 256, 0, stream>>>(P[1], w1 + 1 * 2 * 32, Acc, 0);
  {
    int prev = 0, cur = 1, cq = 1;
    for (int k = 2; k < 5; ++k) {
      int nxt = 3 - cur - prev, nq = 1 - cq;
      k_gemm_cheb<128, 32, 8, 1, 32><<<dim3(1, 64), 256, 0, stream>>>(
          A16, Q[cq], P[prev], 2.f, -1.f, P[nxt], Q[nq]);
      k_mix<2><<<mixBlocks, 256, 0, stream>>>(P[nxt], w1 + k * 2 * 32, Acc, 0);
      prev = cur; cur = nxt; cq = nq;
    }
  }
  k_bias_relu<<<(NDIM * 512) / 256, 256, 0, stream>>>(Acc, b1, P[0], Q[0]);

  // ---------------- layer 2 (F_in = 32, C = 512) ----------------
  k_mix<32><<<mixBlocks, 256, 0, stream>>>(P[0], w2, Acc, 1);
  k_gemm_cheb<128, 128, 4, 2, 512><<<dim3(4, 64), 256, 0, stream>>>(
      A16, Q[0], nullptr, 1.f, 0.f, P[1], Q[1]);
  k_mix<32><<<mixBlocks, 256, 0, stream>>>(P[1], w2 + 1 * 32 * 32, Acc, 0);
  {
    int prev = 0, cur = 1, cq = 1;
    for (int k = 2; k < 5; ++k) {
      int nxt = 3 - cur - prev, nq = 1 - cq;
      k_gemm_cheb<128, 128, 4, 2, 512><<<dim3(4, 64), 256, 0, stream>>>(
          A16, Q[cq], P[prev], 2.f, -1.f, P[nxt], Q[nq]);
      k_mix<32><<<mixBlocks, 256, 0, stream>>>(P[nxt], w2 + k * 32 * 32, Acc, 0);
      prev = cur; cur = nxt; cq = nq;
    }
  }
  k_bias_relu<<<(NDIM * 512) / 256, 256, 0, stream>>>(Acc, b2, Acc, (bf16*)nullptr);

  // ---------------- FC head ----------------
  k_fc1_wmma<<<dim3(32, 64), 32, 0, stream>>>(Acc, fw1, part);
  k_fc1_reduce<<<32, 256, 0, stream>>>(part, fb1, f1o);
  k_fc2<<<8, 256, 0, stream>>>(f1o, fw2, fb2, f2o);
  k_fc3_softmax<<<1, 16, 0, stream>>>(f2o, fw3, fb3, out);
}
